// MultiAttentionBlock_27281632264691
// MI455X (gfx1250) — matrix operand order repaired and checked by reading, not compiled, not run on hardware
//
#include <hip/hip_runtime.h>

// ---------------------------------------------------------------------------
// CDNA5 (gfx1250) multi-head attention block: bf16 WMMA throughout, with
// async global->LDS staging (ASYNCcnt) where the toolchain exposes it.
// ---------------------------------------------------------------------------

typedef __attribute__((ext_vector_type(16))) __bf16 v16bf;
typedef __attribute__((ext_vector_type(8)))  __bf16 v8bf;
typedef __attribute__((ext_vector_type(8)))  float  v8f;
typedef __attribute__((__vector_size__(16))) int    i32x4;

#define DEV static __device__ __forceinline__

#if __has_builtin(__builtin_amdgcn_global_load_async_to_lds_b128)
#define ASYNC_CP 1
#else
#define ASYNC_CP 0
#endif

DEV unsigned short f2bf(float f) {            // f32 -> bf16 (round-nearest-even)
  unsigned u = __float_as_uint(f);
  u += 0x7fffu + ((u >> 16) & 1u);
  return (unsigned short)(u >> 16);
}

union BF16x16 { v16bf v; v8bf h[2]; };
union U128    { uint4 q; unsigned short s[8]; };

// 16B global -> LDS copy; async (no VGPR round-trip, ASYNCcnt) when available.
DEV void ldg128_to_lds(const unsigned short* g, unsigned short* l) {
#if ASYNC_CP
  __builtin_amdgcn_global_load_async_to_lds_b128(
      (__attribute__((address_space(1))) i32x4*)(unsigned short*)g,
      (__attribute__((address_space(3))) i32x4*)l, 0, 0);
#else
  *(uint4*)l = *(const uint4*)g;
#endif
}

DEV void async_wait0() {
#if ASYNC_CP
#if __has_builtin(__builtin_amdgcn_s_wait_asynccnt)
  __builtin_amdgcn_s_wait_asynccnt(0);
#else
  asm volatile("s_wait_asynccnt 0" ::: "memory");
#endif
#endif
}

// Build a v16bf fragment from two 16-byte LDS chunks (ds_load_b128 each).
DEV v16bf ld_frag(const unsigned short* p0, const unsigned short* p1) {
  BF16x16 u;
  u.h[0] = *reinterpret_cast<const v8bf*>(p0);
  u.h[1] = *reinterpret_cast<const v8bf*>(p1);
  return u.v;
}

DEV v8f wmma_bf16(v16bf a, v16bf b, v8f c) {
  return __builtin_amdgcn_wmma_f32_16x16x32_bf16(false, a, false, b,
                                                 (short)0, c, false, false);
}

// ---------------------------------------------------------------------------
// GEMM:  Out[M,N] = A[M,K] * W[N,K]^T + bias[N]
// Workgroup: 256 threads (8 waves), tile 128x128, K-step 32.
// Wave grid 2x4; each wave computes 64x32 (4 M-tiles x 2 N-tiles).
// A may be f32 (converted to bf16 while staging) or bf16; Out bf16 or f32.
// ---------------------------------------------------------------------------
template <bool A_BF16, bool OUT_BF16>
__global__ __launch_bounds__(256) void gemm_nt_kernel(
    const void* __restrict__ Ap, const float* __restrict__ W,
    const float* __restrict__ bias, void* __restrict__ Outp,
    int M, int N, int K)
{
  // Row stride 40 (pad 8) keeps 16B alignment (80B rows) and spreads banks.
  __shared__ __align__(16) unsigned short lx[128][40];
  __shared__ __align__(16) unsigned short lw[128][40];

  const int tid  = threadIdx.x;
  const int lane = tid & 31, wid = tid >> 5;
  const int wm = wid >> 2, wn = wid & 3;       // 2 x 4 wave grid
  const int l16 = lane & 15, half = lane >> 4;
  const int bm = blockIdx.y * 128, bn = blockIdx.x * 128;

  const int row = tid >> 1;                    // 0..127
  const int seg = tid & 1;                     // 16 k-elements per segment

  v8f acc[4][2] = {};

  for (int ko = 0; ko < K; ko += 32) {
    __syncthreads();
    // ---- stage A tile [128 x 32] into LDS as bf16 ----
    if constexpr (A_BF16) {
      const unsigned short* a = (const unsigned short*)Ap +
                                (size_t)(bm + row) * K + ko + seg * 16;
      __builtin_prefetch(a + 32, 0, 1);        // next K-tile (speculative)
      *(uint4*)&lx[row][seg * 16]     = *(const uint4*)(a);
      *(uint4*)&lx[row][seg * 16 + 8] = *(const uint4*)(a + 8);
    } else {
      const float* a = (const float*)Ap + (size_t)(bm + row) * K + ko + seg * 16;
      __builtin_prefetch(a + 32, 0, 1);        // next K-tile (speculative)
      U128 u0, u1;
      #pragma unroll
      for (int e = 0; e < 8; ++e) u0.s[e] = f2bf(a[e]);
      #pragma unroll
      for (int e = 0; e < 8; ++e) u1.s[e] = f2bf(a[8 + e]);
      *(uint4*)&lx[row][seg * 16]     = u0.q;
      *(uint4*)&lx[row][seg * 16 + 8] = u1.q;
    }
    // ---- stage W tile [128 x 32] row-major [n][k] (i.e. B = W^T ready) ----
    {
      const float* w = W + (size_t)(bn + row) * K + ko + seg * 16;
      __builtin_prefetch(w + 32, 0, 1);
      U128 w0, w1;
      #pragma unroll
      for (int e = 0; e < 8; ++e) w0.s[e] = f2bf(w[e]);
      #pragma unroll
      for (int e = 0; e < 8; ++e) w1.s[e] = f2bf(w[8 + e]);
      *(uint4*)&lw[row][seg * 16]     = w0.q;
      *(uint4*)&lw[row][seg * 16 + 8] = w1.q;
    }
    __syncthreads();

    // ---- fragments ----
    v16bf af[4], bfr[2];
    #pragma unroll
    for (int mt = 0; mt < 4; ++mt) {
      // A layout: m = lane&15; K chunks at {half*8, 16+half*8}
      const unsigned short* r0 = &lx[wm * 64 + mt * 16 + l16][half * 8];
      af[mt] = ld_frag(r0, r0 + 16);
    }
    #pragma unroll
    for (int nt = 0; nt < 2; ++nt) {
      // B layout: n = lane&15; K chunks at {half*8, 16+half*8}
      const unsigned short* r0 = &lw[wn * 32 + nt * 16 + l16][half * 8];
      bfr[nt] = ld_frag(r0, r0 + 16);
    }
    #pragma unroll
    for (int mt = 0; mt < 4; ++mt)
      #pragma unroll
      for (int nt = 0; nt < 2; ++nt)
        acc[mt][nt] = wmma_bf16(af[mt], bfr[nt], acc[mt][nt]);
  }

  // ---- epilogue: add bias, write out ----
  #pragma unroll
  for (int nt = 0; nt < 2; ++nt) {
    const int gn = bn + wn * 32 + nt * 16 + l16;
    const float bv = bias[gn];
    #pragma unroll
    for (int mt = 0; mt < 4; ++mt) {
      #pragma unroll
      for (int r = 0; r < 8; ++r) {
        const int gm = bm + wm * 64 + mt * 16 + r + half * 8;  // C/D row map
        const float v = acc[mt][nt][r] + bv;
        if constexpr (OUT_BF16)
          ((unsigned short*)Outp)[(size_t)gm * N + gn] = f2bf(v);
        else
          ((float*)Outp)[(size_t)gm * N + gn] = v;
      }
    }
  }
}

// ---------------------------------------------------------------------------
// Flash attention: one workgroup per (b, h, 128-query block); 8 waves, each
// wave owns 16 query rows. Key dimension swept in 32-key blocks.
// ---------------------------------------------------------------------------
#define S_  2048
#define D_  1024
#define DK_ 64

__global__ __launch_bounds__(256) void attn_kernel(
    const unsigned short* __restrict__ Qp, const unsigned short* __restrict__ Kp,
    const unsigned short* __restrict__ Vp, const int* __restrict__ mask,
    unsigned short* __restrict__ Ctx)
{
  __shared__ __align__(16) unsigned short lq[128][72];   // Q block [q][dk]
  __shared__ __align__(16) unsigned short lk[32][72];    // K block [key][dk]
  __shared__ __align__(16) unsigned short lvt[64][40];   // V block transposed [dk][key]
  __shared__ __align__(16) unsigned short lp[8][16][40]; // per-wave P tile [m][key]

  const int tid = threadIdx.x, lane = tid & 31, wid = tid >> 5;
  const int l16 = lane & 15, half = lane >> 4;
  const int qb = blockIdx.x * 128;
  const int h  = blockIdx.y;
  const int b  = blockIdx.z;
  const size_t headoff = (size_t)b * S_ * D_ + (size_t)h * DK_;

  // ---- stage Q block [128 x 64] once (async -> LDS when available) ----
  {
    const int row = tid >> 1, hseg = tid & 1;
    const unsigned short* src = Qp + headoff + (size_t)(qb + row) * D_ + hseg * 32;
    unsigned short* dst = &lq[row][hseg * 32];
    #pragma unroll
    for (int i = 0; i < 4; ++i) ldg128_to_lds(src + i * 8, dst + i * 8);
  }
  async_wait0();
  __syncthreads();

  // preload Q fragments (reused across all 64 key blocks)
  v16bf aq[2];
  #pragma unroll
  for (int ks = 0; ks < 2; ++ks) {
    const unsigned short* r0 = &lq[wid * 16 + l16][ks * 32 + half * 8];
    aq[ks] = ld_frag(r0, r0 + 16);
  }

  float rmax[8], rsum[8];
  v8f cacc[4] = {};
  #pragma unroll
  for (int r = 0; r < 8; ++r) { rmax[r] = -3.0e38f; rsum[r] = 0.0f; }

  const int* mrow = mask + (size_t)b * S_ * S_;

  for (int kb = 0; kb < S_; kb += 32) {
    __syncthreads();
    // ---- stage K [key][dk] (async) and V transposed [dk][key] ----
    {
      const int r = tid >> 3, c8 = (tid & 7) * 8;
      ldg128_to_lds(Kp + headoff + (size_t)(kb + r) * D_ + c8, &lk[r][c8]);
      U128 u;
      u.q = *(const uint4*)(Vp + headoff + (size_t)(kb + r) * D_ + c8);
      #pragma unroll
      for (int e = 0; e < 8; ++e) lvt[c8 + e][r] = u.s[e];
    }
    async_wait0();
    __syncthreads();

    // ---- scores S = Q K^T : 16x32 tile per wave (two 16x16 N-tiles) ----
    v8f s0 = {}, s1 = {};
    #pragma unroll
    for (int ks = 0; ks < 2; ++ks) {
      const unsigned short* k0 = &lk[l16][ks * 32 + half * 8];
      const unsigned short* k1 = &lk[16 + l16][ks * 32 + half * 8];
      v16bf b0 = ld_frag(k0, k0 + 16);
      v16bf b1 = ld_frag(k1, k1 + 16);
      s0 = wmma_bf16(aq[ks], b0, s0);
      s1 = wmma_bf16(aq[ks], b1, s1);
    }

    // ---- mask + scale + online softmax (rows live in 16-lane halves) ----
    #pragma unroll
    for (int r = 0; r < 8; ++r) {
      const int qg = qb + wid * 16 + r + half * 8;
      const size_t mb = (size_t)qg * S_ + kb;
      float v0 = s0[r] * 0.125f;                 // 1/sqrt(DK)
      float v1 = s1[r] * 0.125f;
      if (mrow[mb + l16] == 0)      v0 = -1.0e9f;
      if (mrow[mb + 16 + l16] == 0) v1 = -1.0e9f;

      float mx = fmaxf(v0, v1);
      mx = fmaxf(mx, __shfl_xor(mx, 1));
      mx = fmaxf(mx, __shfl_xor(mx, 2));
      mx = fmaxf(mx, __shfl_xor(mx, 4));
      mx = fmaxf(mx, __shfl_xor(mx, 8));

      const float nm   = fmaxf(rmax[r], mx);
      const float corr = __expf(rmax[r] - nm);
      rmax[r] = nm;

      const float p0 = __expf(v0 - nm);
      const float p1 = __expf(v1 - nm);
      float ps = p0 + p1;
      ps += __shfl_xor(ps, 1);
      ps += __shfl_xor(ps, 2);
      ps += __shfl_xor(ps, 4);
      ps += __shfl_xor(ps, 8);
      rsum[r] = rsum[r] * corr + ps;

      #pragma unroll
      for (int t = 0; t < 4; ++t) cacc[t][r] *= corr;

      // D-layout -> LDS (A-layout staging for the P*V WMMA)
      const int m = r + half * 8;
      lp[wid][m][l16]      = f2bf(p0);
      lp[wid][m][16 + l16] = f2bf(p1);
    }

    // ---- ctx += P * V ---- (in-wave LDS ordering; no barrier needed)
    const unsigned short* p0 = &lp[wid][l16][half * 8];
    v16bf ap = ld_frag(p0, p0 + 16);
    #pragma unroll
    for (int t = 0; t < 4; ++t) {
      const unsigned short* vr = &lvt[t * 16 + l16][half * 8];
      v16bf bv = ld_frag(vr, vr + 16);
      cacc[t] = wmma_bf16(ap, bv, cacc[t]);
    }
  }

  // ---- epilogue: ctx / l, store bf16 ----
  #pragma unroll
  for (int t = 0; t < 4; ++t)
    #pragma unroll
    for (int r = 0; r < 8; ++r) {
      const int qg = qb + wid * 16 + r + half * 8;
      const float v = cacc[t][r] / rsum[r];
      Ctx[headoff + (size_t)qg * D_ + t * 16 + l16] = f2bf(v);
    }
}

// ---------------------------------------------------------------------------
extern "C" void kernel_launch(void* const* d_in, const int* in_sizes, int n_in,
                              void* d_out, int out_size, void* d_ws, size_t ws_size,
                              hipStream_t stream) {
  (void)in_sizes; (void)n_in; (void)out_size; (void)ws_size;
  const float* q   = (const float*)d_in[0];
  const float* k   = (const float*)d_in[1];
  const float* v   = (const float*)d_in[2];
  const int*   msk = (const int*)d_in[3];
  const float* w_q = (const float*)d_in[4];
  const float* b_q = (const float*)d_in[5];
  const float* w_k = (const float*)d_in[6];
  const float* b_k = (const float*)d_in[7];
  const float* w_v = (const float*)d_in[8];
  const float* b_v = (const float*)d_in[9];
  const float* w_o = (const float*)d_in[10];
  const float* b_o = (const float*)d_in[11];

  const size_t elems = (size_t)2 * 2048 * 1024;          // B*S*D
  unsigned short* qp  = (unsigned short*)d_ws;           // 8 MB each
  unsigned short* kp  = qp + elems;
  unsigned short* vp  = kp + elems;
  unsigned short* ctx = vp + elems;                      // 32 MB total

  dim3 gg(1024 / 128, 4096 / 128);                       // (N tiles, M tiles)
  gemm_nt_kernel<false, true><<<gg, 256, 0, stream>>>(q, w_q, b_q, qp, 4096, 1024, 1024);
  gemm_nt_kernel<false, true><<<gg, 256, 0, stream>>>(k, w_k, b_k, kp, 4096, 1024, 1024);
  gemm_nt_kernel<false, true><<<gg, 256, 0, stream>>>(v, w_v, b_v, vp, 4096, 1024, 1024);

  attn_kernel<<<dim3(2048 / 128, 16, 2), 256, 0, stream>>>(qp, kp, vp, msk, ctx);

  gemm_nt_kernel<true, false><<<gg, 256, 0, stream>>>(ctx, w_o, b_o, d_out, 4096, 1024, 1024);
}
